// GraphPolicyNetwork_32650341384872
// MI455X (gfx1250) — compile-verified, matrix operand order checked by reading, not run on hardware
//
#include <hip/hip_runtime.h>
#include <hip/hip_bf16.h>

// ---------------------------------------------------------------------------
// GCN forward for gfx1250 (MI455X): dense bf16 WMMA GEMMs (8 WMMA / K-step
// per wave, b128 LDS fragment loads, async global->LDS weight staging) +
// bandwidth-bound gather/scatter aggregation + softmax.
// ---------------------------------------------------------------------------

typedef __attribute__((ext_vector_type(16))) __bf16 v16bf;
typedef __attribute__((ext_vector_type(8)))  float  v8f;
typedef int v4i_vec __attribute__((vector_size(16)));   // matches builtin param

#define CH 512          // IN_CH == HID == 512
#define GEMM_BLOCK 256  // 8 waves

#if defined(__gfx1250__) && __has_builtin(__builtin_amdgcn_global_load_async_to_lds_b128)
#define HAVE_ASYNC_LDS 1
#else
#define HAVE_ASYNC_LDS 0
#endif

struct B32x8 { uint4 lo; uint4 hi; };           // 32 bytes == v16bf

union Pack8 { __bf16 h[8]; uint4 v; };          // 8 bf16 == 16 bytes

// ---------------------------------------------------------------------------
// Degree / normalization
// ---------------------------------------------------------------------------
__global__ void init_deg_kernel(float* __restrict__ deg, int n) {
    int i = blockIdx.x * blockDim.x + threadIdx.x;
    if (i < n) deg[i] = 1.0f;   // self-loop contributes 1 to every node
}

__global__ void deg_accum_kernel(const long long* __restrict__ ei,
                                 float* __restrict__ deg, int E) {
    int i = blockIdx.x * blockDim.x + threadIdx.x;
    if (i < E) {
        int d = (int)ei[E + i];       // row 1 = dst
        atomicAdd(deg + d, 1.0f);
    }
}

__global__ void dinv_kernel(const float* __restrict__ deg,
                            float* __restrict__ dinv, int n) {
    int i = blockIdx.x * blockDim.x + threadIdx.x;
    if (i < n) dinv[i] = rsqrtf(deg[i]);   // deg >= 1 always
}

// ---------------------------------------------------------------------------
// Weight pre-pack: Wp[n][k] = bf16(W[k][n])  (column-major bf16, 512 KB,
// L2-resident; packed once per layer, reused by every GEMM block)
// ---------------------------------------------------------------------------
__global__ void pack_w_kernel(const float* __restrict__ W,
                              __bf16* __restrict__ Wp) {
    int idx = blockIdx.x * blockDim.x + threadIdx.x;   // 512*512 threads
    int k = idx >> 9;
    int n = idx & (CH - 1);
    Wp[(size_t)n * CH + k] = (__bf16)W[(size_t)k * CH + n];
}

// ---------------------------------------------------------------------------
// C[M x 512] = A[M x 512] * W[512 x 512]   (W given pre-packed as Wp[n][k])
//
// Block: 256 threads = 8 waves in a 2(M) x 4(N) grid of wave tiles.
// Wave tile: 32(M) x 64(N) = 2 x 4 subtiles of 16x16 -> 8 v8f accumulators,
// 8 v_wmma_f32_16x16x32_bf16 per K-step per wave.
// Block tile: 64(M) x 256(N). Grid: (512/256, ceil(M/64)).
// Fragment layout (ISA 7.12.2, 16-bit): lane holds K runs [kb..kb+7] and
// [kb+16..kb+23] (kb = 0|8) -> two ds_load_b128 per fragment.
// ---------------------------------------------------------------------------
__global__ void gemm512_bf16_kernel(const float* __restrict__ A,
                                    __bf16* __restrict__ Wp,
                                    float* __restrict__ C, int M) {
    // row stride 40 bf16 = 80B: keeps the 4 16B runs aligned, staggers banks
    __shared__ __bf16 Alds[64][40];    // 64 x 32 A tile (f32 -> bf16)
    __shared__ __bf16 Blds[256][40];   // 256(N) x 32(K), K-contiguous rows

    const int tid  = threadIdx.x;
    const int lane = tid & 31;
    const int wave = tid >> 5;
    const int wm   = wave >> 2;        // 0..1
    const int wn   = wave & 3;         // 0..3
    const int row0 = blockIdx.y * 64;
    const int nbase = blockIdx.x * 256;

    v8f acc[2][4];
#pragma unroll
    for (int i = 0; i < 2; ++i)
#pragma unroll
        for (int j = 0; j < 4; ++j)
            acc[i][j] = v8f{0.f, 0.f, 0.f, 0.f, 0.f, 0.f, 0.f, 0.f};

    for (int k0 = 0; k0 < CH; k0 += 32) {
        // ---- stage B tile: 256 rows x 32 K of bf16, pure 16B copies ----
        {
            const int t4 = tid & 3;            // 16B chunk within a row
            const int rb = tid >> 2;           // 0..63
#pragma unroll
            for (int pass = 0; pass < 4; ++pass) {
                int n = rb + pass * 64;        // 0..255
                __bf16* gsrc = Wp + ((size_t)(nbase + n) * CH + k0 + t4 * 8);
#if HAVE_ASYNC_LDS
                __builtin_amdgcn_global_load_async_to_lds_b128(
                    (v4i_vec*)gsrc, (v4i_vec*)&Blds[n][t4 * 8], 0, 0);
#else
                *(uint4*)&Blds[n][t4 * 8] = *(const uint4*)gsrc;
#endif
            }
        }
        // ---- stage A tile: 64 rows x 32 K, f32 -> bf16, 8 elems/thread ----
        {
            int r  = tid >> 2;                 // 0..63
            int cb = (tid & 3) * 8;            // 0,8,16,24
            int grow = row0 + r;
            if (grow > M - 1) grow = M - 1;    // clamp tail rows
            const float* ap = A + (size_t)grow * CH + k0 + cb;
            float4 f0 = *(const float4*)ap;
            float4 f1 = *(const float4*)(ap + 4);
            Pack8 p;
            p.h[0] = (__bf16)f0.x; p.h[1] = (__bf16)f0.y;
            p.h[2] = (__bf16)f0.z; p.h[3] = (__bf16)f0.w;
            p.h[4] = (__bf16)f1.x; p.h[5] = (__bf16)f1.y;
            p.h[6] = (__bf16)f1.z; p.h[7] = (__bf16)f1.w;
            *(uint4*)&Alds[r][cb] = p.v;
        }
#if HAVE_ASYNC_LDS
#if __has_builtin(__builtin_amdgcn_s_wait_asynccnt)
        __builtin_amdgcn_s_wait_asynccnt(0);
#else
        asm volatile("s_wait_asynccnt 0x0" ::: "memory");
#endif
#endif
        __syncthreads();

        // ---- fragments: two ds_load_b128 each ----
        const int lm = lane & 15;
        const int kb = (lane & 16) ? 8 : 0;

        v16bf afrag[2];
#pragma unroll
        for (int sm = 0; sm < 2; ++sm) {
            int row = wm * 32 + sm * 16 + lm;
            B32x8 r;
            r.lo = *(const uint4*)&Alds[row][kb];
            r.hi = *(const uint4*)&Alds[row][kb + 16];
            afrag[sm] = __builtin_bit_cast(v16bf, r);
        }
        v16bf bfrag[4];
#pragma unroll
        for (int sn = 0; sn < 4; ++sn) {
            int nr = wn * 64 + sn * 16 + lm;
            B32x8 r;
            r.lo = *(const uint4*)&Blds[nr][kb];
            r.hi = *(const uint4*)&Blds[nr][kb + 16];
            bfrag[sn] = __builtin_bit_cast(v16bf, r);
        }

        // ---- 8 WMMAs ----
#pragma unroll
        for (int sm = 0; sm < 2; ++sm)
#pragma unroll
            for (int sn = 0; sn < 4; ++sn)
                acc[sm][sn] = __builtin_amdgcn_wmma_f32_16x16x32_bf16(
                    false, afrag[sm], false, bfrag[sn],
                    (short)0, acc[sm][sn], false, false);

        __syncthreads();
    }

    // ---- store C (guard tail rows; N always in range) ----
    const int lm   = lane & 15;
    const int mhalf = (lane & 16) ? 8 : 0;
#pragma unroll
    for (int sm = 0; sm < 2; ++sm) {
        int mbase = row0 + wm * 32 + sm * 16 + mhalf;
#pragma unroll
        for (int sn = 0; sn < 4; ++sn) {
            int ncol = nbase + wn * 64 + sn * 16 + lm;
#pragma unroll
            for (int r = 0; r < 8; ++r) {
                int mr = mbase + r;
                if (mr < M) C[(size_t)mr * CH + ncol] = acc[sm][sn][r];
            }
        }
    }
}

// ---------------------------------------------------------------------------
// Aggregation: out[i][c] = h[i][c] * dinv[i]^2  (self-loop term; also inits)
// ---------------------------------------------------------------------------
__global__ void selfloop_kernel(const float* __restrict__ h,
                                const float* __restrict__ dinv,
                                float* __restrict__ out, int n) {
    size_t idx = (size_t)blockIdx.x * blockDim.x + threadIdx.x;
    size_t total = (size_t)n * CH;
    if (idx < total) {
        int node = (int)(idx >> 9);          // /512
        float s = dinv[node];
        out[idx] = h[idx] * s * s;
    }
}

// one wave per edge; float4 gather + atomic scatter-add
__global__ void edge_scatter_kernel(const float* __restrict__ h,
                                    const long long* __restrict__ ei,
                                    const float* __restrict__ dinv,
                                    float* __restrict__ out, int E) {
    int gid  = blockIdx.x * blockDim.x + threadIdx.x;
    int edge = gid >> 5;
    int lane = gid & 31;
    if (edge >= E) return;
    int s = (int)ei[edge];          // row 0 = src
    int d = (int)ei[E + edge];      // row 1 = dst
    float norm = dinv[s] * dinv[d];
    const float* hs = h + (size_t)s * CH;
    float* od = out + (size_t)d * CH;
#pragma unroll
    for (int j = 0; j < 4; ++j) {
        int c = j * 128 + lane * 4;
        float4 v = *(const float4*)(hs + c);
        atomicAdd(od + c + 0, v.x * norm);
        atomicAdd(od + c + 1, v.y * norm);
        atomicAdd(od + c + 2, v.z * norm);
        atomicAdd(od + c + 3, v.w * norm);
    }
}

__global__ void bias_relu_kernel(float* __restrict__ out,
                                 const float* __restrict__ b, int n) {
    size_t idx = (size_t)blockIdx.x * blockDim.x + threadIdx.x;
    size_t total = (size_t)n * CH;
    if (idx < total) {
        int c = (int)(idx & (CH - 1));
        float v = out[idx] + b[c];
        out[idx] = v > 0.f ? v : 0.f;
    }
}

// ---------------------------------------------------------------------------
// logits[i] = h[i] . Wo + bo  (one wave per row, shuffle reduction)
// ---------------------------------------------------------------------------
__global__ void logits_kernel(const float* __restrict__ h,
                              const float* __restrict__ Wo,
                              const float* __restrict__ bo,
                              float* __restrict__ logits, int n) {
    int gid  = blockIdx.x * blockDim.x + threadIdx.x;
    int row  = gid >> 5;
    int lane = gid & 31;
    if (row >= n) return;
    const float* hr = h + (size_t)row * CH;
    float s = 0.f;
#pragma unroll
    for (int j = 0; j < CH / 32; ++j) {
        int c = j * 32 + lane;
        s += hr[c] * Wo[c];
    }
#pragma unroll
    for (int off = 16; off > 0; off >>= 1) s += __shfl_down(s, off, 32);
    if (lane == 0) logits[row] = s + bo[0];
}

// ---------------------------------------------------------------------------
// Softmax over n logits: pass 1 (single block): max + sum(exp); pass 2: norm.
// ---------------------------------------------------------------------------
__global__ void softmax_reduce_kernel(const float* __restrict__ l,
                                      float* __restrict__ red, int n) {
    __shared__ float sm[1024];
    int tid = threadIdx.x;
    float m = -3.402823466e38f;
    for (int i = tid; i < n; i += blockDim.x) m = fmaxf(m, l[i]);
    sm[tid] = m;
    __syncthreads();
    for (int s = 512; s > 0; s >>= 1) {
        if (tid < s) sm[tid] = fmaxf(sm[tid], sm[tid + s]);
        __syncthreads();
    }
    float gmax = sm[0];
    __syncthreads();
    float sum = 0.f;
    for (int i = tid; i < n; i += blockDim.x) sum += __expf(l[i] - gmax);
    sm[tid] = sum;
    __syncthreads();
    for (int s = 512; s > 0; s >>= 1) {
        if (tid < s) sm[tid] += sm[tid + s];
        __syncthreads();
    }
    if (tid == 0) { red[0] = gmax; red[1] = sm[0]; }
}

__global__ void softmax_final_kernel(const float* __restrict__ l,
                                     const float* __restrict__ red,
                                     float* __restrict__ out, int n) {
    int i = blockIdx.x * blockDim.x + threadIdx.x;
    if (i < n) out[i] = __expf(l[i] - red[0]) / red[1];
}

// ---------------------------------------------------------------------------
// Host-side orchestration
// ---------------------------------------------------------------------------
extern "C" void kernel_launch(void* const* d_in, const int* in_sizes, int n_in,
                              void* d_out, int out_size, void* d_ws, size_t ws_size,
                              hipStream_t stream) {
    const float*      x  = (const float*)d_in[0];
    const long long*  ei = (const long long*)d_in[1];   // int64 [2, E]
    const float*      W1 = (const float*)d_in[2];
    const float*      b1 = (const float*)d_in[3];
    const float*      W2 = (const float*)d_in[4];
    const float*      b2 = (const float*)d_in[5];
    const float*      Wo = (const float*)d_in[6];
    const float*      bo = (const float*)d_in[7];
    float*            out = (float*)d_out;

    const int N = in_sizes[0] / CH;   // 50000
    const int E = in_sizes[1] / 2;    // 800000

    // workspace layout (floats)
    float* ws     = (float*)d_ws;
    float* deg    = ws;                       // N
    float* dinv   = ws + N;                   // N
    float* logits = ws + 2 * (size_t)N;       // N
    float* red    = ws + 3 * (size_t)N;       // 2
    float* bufA   = ws + 3 * (size_t)N + 64;  // N*512
    float* bufB   = bufA + (size_t)N * CH;    // N*512
    __bf16* Wp    = (__bf16*)(bufB + (size_t)N * CH);   // 512*512 bf16

    const int T = 256;
    const int nBlkN   = (N + T - 1) / T;
    const int nBlkE   = (E + T - 1) / T;
    const int nBlkNE  = (E * 32 + T - 1) / T;            // wave per edge
    const int nBlkNC  = (int)(((size_t)N * CH + T - 1) / T);
    const int nBlkRow = (N * 32 + T - 1) / T;            // wave per row
    const int nBlkW   = (CH * CH) / T;                   // 1024
    dim3 gemmGrid(CH / 256, (N + 63) / 64);              // (2, 782)

    // normalization
    init_deg_kernel<<<nBlkN, T, 0, stream>>>(deg, N);
    deg_accum_kernel<<<nBlkE, T, 0, stream>>>(ei, deg, E);
    dinv_kernel<<<nBlkN, T, 0, stream>>>(deg, dinv, N);

    // layer 1: h = relu(aggregate(x @ W1) + b1)
    pack_w_kernel<<<nBlkW, T, 0, stream>>>(W1, Wp);
    gemm512_bf16_kernel<<<gemmGrid, GEMM_BLOCK, 0, stream>>>(x, Wp, bufA, N);
    selfloop_kernel<<<nBlkNC, T, 0, stream>>>(bufA, dinv, bufB, N);
    edge_scatter_kernel<<<nBlkNE, T, 0, stream>>>(bufA, ei, dinv, bufB, E);
    bias_relu_kernel<<<nBlkNC, T, 0, stream>>>(bufB, b1, N);

    // layer 2: h = relu(aggregate(h @ W2) + b2)
    pack_w_kernel<<<nBlkW, T, 0, stream>>>(W2, Wp);
    gemm512_bf16_kernel<<<gemmGrid, GEMM_BLOCK, 0, stream>>>(bufB, Wp, bufA, N);
    selfloop_kernel<<<nBlkNC, T, 0, stream>>>(bufA, dinv, bufB, N);
    edge_scatter_kernel<<<nBlkNE, T, 0, stream>>>(bufA, ei, dinv, bufB, E);
    bias_relu_kernel<<<nBlkNC, T, 0, stream>>>(bufB, b2, N);

    // output head + softmax
    logits_kernel<<<nBlkRow, T, 0, stream>>>(bufB, Wo, bo, logits, N);
    softmax_reduce_kernel<<<1, 1024, 0, stream>>>(logits, red, N);
    softmax_final_kernel<<<nBlkN, T, 0, stream>>>(logits, red, out, N);
}
